// Experts_38165079392793
// MI455X (gfx1250) — matrix operand order repaired and checked by reading, not compiled, not run on hardware
//
#include <hip/hip_runtime.h>
#include <hip/hip_bf16.h>

#define HDIM 1024
#define IDIM 1024
#define NEXP 8
#define TMAX 2048

typedef __bf16 bf16_t;
typedef __attribute__((ext_vector_type(8)))  __bf16 v8bf;
typedef __attribute__((ext_vector_type(16))) __bf16 v16bf;
typedef __attribute__((ext_vector_type(8)))  float  v8f;

#if defined(__AMDGCN__) && __has_builtin(__builtin_amdgcn_global_load_async_to_lds_b128)
#define HAVE_ASYNC_LDS 1
#else
#define HAVE_ASYNC_LDS 0
#endif

// 128-bit payload type matching the async-LDS builtin's prototype
typedef int v4i_t __attribute__((vector_size(16)));
typedef __attribute__((address_space(1))) v4i_t* gptr128_t;
typedef __attribute__((address_space(3))) v4i_t* lptr128_t;

// fp32 -> bf16 round-to-nearest-even
__device__ __forceinline__ bf16_t f2bf(float f) {
  unsigned u = __builtin_bit_cast(unsigned, f);
  u += 0x7FFFu + ((u >> 16) & 1u);
  unsigned short s = (unsigned short)(u >> 16);
  return __builtin_bit_cast(bf16_t, s);
}

__device__ __forceinline__ v16bf cat16(v8bf lo, v8bf hi) {
  return __builtin_shufflevector(lo, hi, 0,1,2,3,4,5,6,7,8,9,10,11,12,13,14,15);
}

// 16B async copy global(bf16, already packed) -> LDS, ASYNCcnt-tracked.
__device__ __forceinline__ void copy16_g2l(const bf16_t* g, bf16_t* l) {
#if HAVE_ASYNC_LDS
  __builtin_amdgcn_global_load_async_to_lds_b128((gptr128_t)g, (lptr128_t)l, 0, 0);
#else
  *(v8bf*)l = *(const v8bf*)g;
#endif
}

__device__ __forceinline__ void wait_async() {
#if HAVE_ASYNC_LDS
#if __has_builtin(__builtin_amdgcn_s_wait_asynccnt)
  __builtin_amdgcn_s_wait_asynccnt(0);
#else
  asm volatile("s_wait_asynccnt 0x0" ::: "memory");
#endif
#endif
}

// A fragment (16x32 bf16, MxK): lane<16 -> M=lane, K = {0..7,16..23};
// lane>=16 -> M=lane-16, K = {8..15,24..31}. Tile in LDS row-major, stride ld.
__device__ __forceinline__ v16bf load_a_frag(const bf16_t* t, int row0, int ld, int lane) {
  int r  = row0 + (lane & 15);
  int kb = (lane >> 4) << 3;              // 0 or 8
  const bf16_t* p = t + r * ld + kb;
  v8bf lo = *(const v8bf*)(p);
  v8bf hi = *(const v8bf*)(p + 16);
  return cat16(lo, hi);
}

// B fragment (32x16 bf16, KxN), stored as Wt tile [n][k] row-major:
// lane<16 -> N=lane, K = {0..7,16..23}; lane>=16 -> N=lane-16, K = {8..15,24..31}.
__device__ __forceinline__ v16bf load_b_frag(const bf16_t* t, int col0, int ld, int lane) {
  int n  = col0 + (lane & 15);
  int kb = (lane >> 4) << 3;              // 0 or 8
  const bf16_t* p = t + n * ld + kb;
  v8bf lo = *(const v8bf*)(p);
  v8bf hi = *(const v8bf*)(p + 16);
  return cat16(lo, hi);
}

__device__ __forceinline__ v8f wmma_bf16(v16bf a, v16bf b, v8f c) {
  return __builtin_amdgcn_wmma_f32_16x16x32_bf16(false, a, false, b, (short)0, c, false, false);
}

// ---------------- init: zero out + counters, h -> bf16 ----------------
__global__ void k_init(const float* __restrict__ h, bf16_t* __restrict__ hbf,
                       float* __restrict__ out, int* __restrict__ cnt, int n) {
  int i = blockIdx.x * blockDim.x + threadIdx.x;
  if (i < n) { out[i] = 0.0f; hbf[i] = f2bf(h[i]); }
  if (i < NEXP) cnt[i] = 0;
}

// ---------------- router: logits, top-2, softmax, build lists ----------------
__global__ __launch_bounds__(64) void k_router(const float* __restrict__ h,
                                               const float* __restrict__ rw,
                                               int* __restrict__ cnt,
                                               int* __restrict__ list,
                                               float* __restrict__ wl) {
  const int tok = blockIdx.x;
  const int tid = threadIdx.x;
  float acc[NEXP];
#pragma unroll
  for (int e = 0; e < NEXP; ++e) acc[e] = 0.f;
  const float* hp = h + (size_t)tok * HDIM;
  for (int i = tid; i < HDIM; i += 64) {
    float x = hp[i];
#pragma unroll
    for (int e = 0; e < NEXP; ++e) acc[e] += x * rw[e * HDIM + i];
  }
  __shared__ float red[64][NEXP];
#pragma unroll
  for (int e = 0; e < NEXP; ++e) red[tid][e] = acc[e];
  __syncthreads();
  if (tid < NEXP) {
    float s = 0.f;
    for (int j = 0; j < 64; ++j) s += red[j][tid];
    red[0][tid] = s;
  }
  __syncthreads();
  if (tid == 0) {
    float l[NEXP];
#pragma unroll
    for (int e = 0; e < NEXP; ++e) l[e] = red[0][e];
    int e0 = 0; float m0 = l[0];
#pragma unroll
    for (int e = 1; e < NEXP; ++e) if (l[e] > m0) { m0 = l[e]; e0 = e; }
    int e1 = -1; float m1 = -3.4e38f;
#pragma unroll
    for (int e = 0; e < NEXP; ++e) if (e != e0 && l[e] > m1) { m1 = l[e]; e1 = e; }
    float t  = __expf(m1 - m0);
    float p0 = 1.f / (1.f + t);
    float p1 = 1.f - p0;
    int s0 = atomicAdd(&cnt[e0], 1);
    list[e0 * TMAX + s0] = tok; wl[e0 * TMAX + s0] = p0;
    int s1 = atomicAdd(&cnt[e1], 1);
    list[e1 * TMAX + s1] = tok; wl[e1 * TMAX + s1] = p1;
  }
}

// ---------------- GEMM1: gathered h @ w1[e]^T, fused silu(gate)*up ----------------
// block tile: M=128 tokens x (64 gate cols + 64 up cols), K-step 32.
#define LDA 40   // padded LDS row stride (elems): conflict-free 16-lane access
__global__ __launch_bounds__(256) void k_gemm1(const bf16_t* __restrict__ hbf,
                                               const float* __restrict__ w1,
                                               const int* __restrict__ cnt,
                                               const int* __restrict__ list,
                                               bf16_t* __restrict__ inter) {
  const int e  = blockIdx.z;
  const int n0 = blockIdx.x * 64;
  const int m0 = blockIdx.y * 128;
  const int c  = cnt[e];
  if (m0 >= c) return;
  int base = 0;
  for (int i = 0; i < e; ++i) base += cnt[i];

  __shared__ bf16_t As [128 * LDA];
  __shared__ bf16_t BsG[ 64 * LDA];
  __shared__ bf16_t BsU[ 64 * LDA];

  const int tid  = threadIdx.x;
  const int lane = tid & 31;
  const int wid  = tid >> 5;
  const int wm   = (wid & 3) * 32;   // 4 wave-rows  x 32
  const int wn   = (wid >> 2) * 32;  // 2 wave-cols  x 32 (per half)

  v8f accG[2][2], accU[2][2];
#pragma unroll
  for (int i = 0; i < 2; ++i)
#pragma unroll
    for (int j = 0; j < 2; ++j) { accG[i][j] = (v8f){0,0,0,0,0,0,0,0}; accU[i][j] = (v8f){0,0,0,0,0,0,0,0}; }

  const int* tl = list + e * TMAX;
  const size_t w1e = (size_t)e * (2 * IDIM) * HDIM;

  // per-thread staging coordinates
  const int arow0 = (tid * 2) >> 2;          // A chunk 0 row
  const int acol0 = ((tid * 2) & 3) * 8;
  const int arow1 = (tid * 2 + 1) >> 2;      // A chunk 1 row
  const int acol1 = ((tid * 2 + 1) & 3) * 8;
  const int brow  = tid >> 2;                // 0..63
  const int bcol  = (tid & 3) * 8;

  int am0 = m0 + arow0; if (am0 >= c) am0 = c - 1;
  int am1 = m0 + arow1; if (am1 >= c) am1 = c - 1;
  const int atok0 = tl[am0];
  const int atok1 = tl[am1];

  for (int k0 = 0; k0 < HDIM; k0 += 32) {
    // stage A (bf16): async global -> LDS, no VGPR bounce
    copy16_g2l(hbf + (size_t)atok0 * HDIM + k0 + acol0, As + arow0 * LDA + acol0);
    copy16_g2l(hbf + (size_t)atok1 * HDIM + k0 + acol1, As + arow1 * LDA + acol1);
    // stage B gate/up: 64x32 each, fp32 -> bf16 convert in flight
    {
      const float* pg = w1 + w1e + (size_t)(n0 + brow)        * HDIM + k0 + bcol;
      const float* pu = w1 + w1e + (size_t)(IDIM + n0 + brow) * HDIM + k0 + bcol;
      if (k0 + 32 < HDIM) {                 // prime L2->WGP for next K tile
        __builtin_prefetch(pg + 32, 0, 3);
        __builtin_prefetch(pu + 32, 0, 3);
      }
      v8bf g, u;
#pragma unroll
      for (int j = 0; j < 8; ++j) { g[j] = f2bf(pg[j]); u[j] = f2bf(pu[j]); }
      *(v8bf*)(BsG + brow * LDA + bcol) = g;
      *(v8bf*)(BsU + brow * LDA + bcol) = u;
    }
    wait_async();
    __syncthreads();

    v16bf a0  = load_a_frag(As,  wm,      LDA, lane);
    v16bf a1  = load_a_frag(As,  wm + 16, LDA, lane);
    v16bf bg0 = load_b_frag(BsG, wn,      LDA, lane);
    v16bf bg1 = load_b_frag(BsG, wn + 16, LDA, lane);
    v16bf bu0 = load_b_frag(BsU, wn,      LDA, lane);
    v16bf bu1 = load_b_frag(BsU, wn + 16, LDA, lane);
    accG[0][0] = wmma_bf16(a0, bg0, accG[0][0]);
    accG[0][1] = wmma_bf16(a0, bg1, accG[0][1]);
    accG[1][0] = wmma_bf16(a1, bg0, accG[1][0]);
    accG[1][1] = wmma_bf16(a1, bg1, accG[1][1]);
    accU[0][0] = wmma_bf16(a0, bu0, accU[0][0]);
    accU[0][1] = wmma_bf16(a0, bu1, accU[0][1]);
    accU[1][0] = wmma_bf16(a1, bu0, accU[1][0]);
    accU[1][1] = wmma_bf16(a1, bu1, accU[1][1]);
    __syncthreads();
  }

  // epilogue: inter = silu(gate) * up  (bf16); fast rcp instead of IEEE divide
#pragma unroll
  for (int mt = 0; mt < 2; ++mt) {
#pragma unroll
    for (int nt = 0; nt < 2; ++nt) {
#pragma unroll
      for (int r = 0; r < 8; ++r) {
        int row = wm + mt * 16 + ((lane >> 4) << 3) + r;
        int m   = m0 + row;
        if (m < c) {
          int col  = n0 + wn + nt * 16 + (lane & 15);
          float g  = accG[mt][nt][r];
          float u  = accU[mt][nt][r];
          float val = g * u * __builtin_amdgcn_rcpf(1.f + __expf(-g));
          inter[(size_t)(base + m) * IDIM + col] = f2bf(val);
        }
      }
    }
  }
}

// ---------------- GEMM2: inter @ w2[e]^T, scaled scatter-add ----------------
// block tile: M=128 x N=128, K-step 32
__global__ __launch_bounds__(256) void k_gemm2(const bf16_t* __restrict__ inter,
                                               const float* __restrict__ w2,
                                               const int* __restrict__ cnt,
                                               const int* __restrict__ list,
                                               const float* __restrict__ wl,
                                               float* __restrict__ out) {
  const int e  = blockIdx.z;
  const int n0 = blockIdx.x * 128;
  const int m0 = blockIdx.y * 128;
  const int c  = cnt[e];
  if (m0 >= c) return;
  int base = 0;
  for (int i = 0; i < e; ++i) base += cnt[i];

  __shared__ bf16_t As[128 * LDA];
  __shared__ bf16_t Bs[128 * LDA];

  const int tid  = threadIdx.x;
  const int lane = tid & 31;
  const int wid  = tid >> 5;
  const int wm   = (wid & 3) * 32;   // 4 wave-rows x 32
  const int wn   = (wid >> 2) * 64;  // 2 wave-cols x 64

  v8f acc[2][4];
#pragma unroll
  for (int i = 0; i < 2; ++i)
#pragma unroll
    for (int j = 0; j < 4; ++j) acc[i][j] = (v8f){0,0,0,0,0,0,0,0};

  const size_t w2e = (size_t)e * HDIM * IDIM;

  const int arow0 = (tid * 2) >> 2;
  const int acol0 = ((tid * 2) & 3) * 8;
  const int arow1 = (tid * 2 + 1) >> 2;
  const int acol1 = ((tid * 2 + 1) & 3) * 8;
  int am0 = m0 + arow0; if (am0 >= c) am0 = c - 1;
  int am1 = m0 + arow1; if (am1 >= c) am1 = c - 1;
  const size_t arow0g = (size_t)(base + am0) * IDIM;
  const size_t arow1g = (size_t)(base + am1) * IDIM;

  for (int k0 = 0; k0 < IDIM; k0 += 32) {
    // stage A (bf16 inter rows): async global -> LDS
    copy16_g2l(inter + arow0g + k0 + acol0, As + arow0 * LDA + acol0);
    copy16_g2l(inter + arow1g + k0 + acol1, As + arow1 * LDA + acol1);
    // stage B: w2 rows, fp32 -> bf16 (two chunks per thread: rows 0..127)
#pragma unroll
    for (int t = 0; t < 2; ++t) {
      int q   = tid * 2 + t;        // 0..511
      int row = q >> 2;             // 0..127
      int col = (q & 3) * 8;
      const float* pb = w2 + w2e + (size_t)(n0 + row) * IDIM + k0 + col;
      if (k0 + 32 < IDIM) __builtin_prefetch(pb + 32, 0, 3);
      v8bf b;
#pragma unroll
      for (int j = 0; j < 8; ++j) b[j] = f2bf(pb[j]);
      *(v8bf*)(Bs + row * LDA + col) = b;
    }
    wait_async();
    __syncthreads();

    v16bf a0 = load_a_frag(As, wm,      LDA, lane);
    v16bf a1 = load_a_frag(As, wm + 16, LDA, lane);
#pragma unroll
    for (int nt = 0; nt < 4; ++nt) {
      v16bf b = load_b_frag(Bs, wn + nt * 16, LDA, lane);
      acc[0][nt] = wmma_bf16(a0, b, acc[0][nt]);
      acc[1][nt] = wmma_bf16(a1, b, acc[1][nt]);
    }
    __syncthreads();
  }

  // scatter: out[tok] += weight * acc
#pragma unroll
  for (int mt = 0; mt < 2; ++mt) {
    int rbase = wm + mt * 16 + ((lane >> 4) << 3);
#pragma unroll
    for (int r = 0; r < 8; ++r) {
      int m = m0 + rbase + r;
      if (m < c) {
        int   tok = list[e * TMAX + m];
        float wgt = wl[e * TMAX + m];
#pragma unroll
        for (int nt = 0; nt < 4; ++nt) {
          int col = n0 + wn + nt * 16 + (lane & 15);
          atomicAdd(&out[(size_t)tok * HDIM + col], wgt * acc[mt][nt][r]);
        }
      }
    }
  }
}

// ---------------- launch ----------------
extern "C" void kernel_launch(void* const* d_in, const int* in_sizes, int n_in,
                              void* d_out, int out_size, void* d_ws, size_t ws_size,
                              hipStream_t stream) {
  const float* h  = (const float*)d_in[0];   // (2,1024,H) fp32
  const float* rw = (const float*)d_in[1];   // (E,H)
  const float* w1 = (const float*)d_in[2];   // (E,2I,H)
  const float* w2 = (const float*)d_in[3];   // (E,H,I)
  float* out = (float*)d_out;

  // workspace layout (~12.2 MB total)
  char* ws = (char*)d_ws;
  int*    cnt   = (int*)   (ws);                                   // 8 ints
  float*  wl    = (float*) (ws + 4096);                            // 8*TMAX f32
  int*    list  = (int*)   (ws + 4096 + 65536);                    // 8*TMAX i32
  bf16_t* hbf   = (bf16_t*)(ws + 4096 + 2 * 65536);                // TMAX*H bf16
  bf16_t* inter = (bf16_t*)(ws + 4096 + 2 * 65536 + (size_t)TMAX * HDIM * 2); // 2*TMAX*I bf16

  const int n = TMAX * HDIM;
  k_init  <<<dim3((n + 255) / 256), dim3(256), 0, stream>>>(h, hbf, out, cnt, n);
  k_router<<<dim3(TMAX),            dim3(64),  0, stream>>>(h, rw, cnt, list, wl);
  k_gemm1 <<<dim3(IDIM / 64,  TMAX / 128, NEXP), dim3(256), 0, stream>>>(hbf, w1, cnt, list, inter);
  k_gemm2 <<<dim3(HDIM / 128, TMAX / 128, NEXP), dim3(256), 0, stream>>>(inter, w2, cnt, list, wl, out);
}
